// CRCLoss_46162308497713
// MI455X (gfx1250) — matrix operand order repaired and checked by reading, not compiled, not run on hardware
//
#include <hip/hip_runtime.h>
#include <hip/hip_bf16.h>

typedef __attribute__((ext_vector_type(16))) _Float16 v16h;
typedef __attribute__((ext_vector_type(8)))  _Float16 v8h;
typedef __attribute__((ext_vector_type(4)))  _Float16 v4h;
typedef __attribute__((ext_vector_type(8)))  float    v8f;

#define BDIM 8192
#define KDIM 1024
#define TEMP_INV 10.0f

#define LDS_STRIDE 40   // halves per row (80 B: 16B-aligned, bank-spread)

// ---------------------------------------------------------------------------
// Kernel 1: single-block scan over labels.
//   perm[i] = slot after stable partition (normals first), nN = #normals.
//   Also zeroes the double accumulators (ws is poisoned, not re-zeroed).
// ---------------------------------------------------------------------------
__global__ __launch_bounds__(1024) void crc_scan(const int* __restrict__ labels,
                                                 int* __restrict__ perm,
                                                 int* __restrict__ nNout,
                                                 double* __restrict__ Sacc,
                                                 double* __restrict__ Lacc) {
    __shared__ int sc[1024];
    const int t = threadIdx.x;
    const int base = t * 8;
    int loc[8];
    int cnt = 0;
    #pragma unroll
    for (int k = 0; k < 8; ++k) {
        loc[k] = (labels[base + k] == 0) ? 1 : 0;
        cnt += loc[k];
    }
    sc[t] = cnt;
    __syncthreads();
    for (int off = 1; off < 1024; off <<= 1) {
        int v = sc[t];
        int add = (t >= off) ? sc[t - off] : 0;
        __syncthreads();
        sc[t] = v + add;
        __syncthreads();
    }
    const int nN = sc[1023];
    int run = sc[t] - cnt;
    #pragma unroll
    for (int k = 0; k < 8; ++k) {
        const int i = base + k;
        perm[i] = loc[k] ? run : (nN + (i - run));
        run += loc[k];
    }
    if (t == 0) { *nNout = nN; *Sacc = 0.0; *Lacc = 0.0; }
}

// ---------------------------------------------------------------------------
// Kernel 2: L2-normalize each row, convert to f16, scatter to permuted slot.
// ---------------------------------------------------------------------------
__global__ __launch_bounds__(256) void crc_norm(const float* __restrict__ F,
                                                const int* __restrict__ perm,
                                                _Float16* __restrict__ fH) {
    const int row  = blockIdx.x;
    const int t    = threadIdx.x;
    const int wave = t >> 5;
    const int lane = t & 31;
    const float4 v = ((const float4*)(F + (size_t)row * KDIM))[t];
    float ss = v.x * v.x + v.y * v.y + v.z * v.z + v.w * v.w;
    #pragma unroll
    for (int off = 16; off >= 1; off >>= 1) ss += __shfl_xor(ss, off, 32);
    __shared__ float red[8];
    if (lane == 0) red[wave] = ss;
    __syncthreads();
    float tot = 0.f;
    #pragma unroll
    for (int w = 0; w < 8; ++w) tot += red[w];
    const float rinv = rsqrtf(tot);
    const int p = perm[row];
    v4h o;
    o[0] = (_Float16)(v.x * rinv);
    o[1] = (_Float16)(v.y * rinv);
    o[2] = (_Float16)(v.z * rinv);
    o[3] = (_Float16)(v.w * rinv);
    *(v4h*)(fH + (size_t)p * KDIM + t * 4) = o;
}

// ---------------------------------------------------------------------------
// Kernels 3/4: LDS-staged WMMA Gram-matrix phases.
//   Block = 8 waves = 2x4 wave grid over a 128x256 block tile.
//   Each wave: 64x64 macro-tile = 4x4 v_wmma_f32_16x16x32_f16 per k-chunk.
//   Double-buffered LDS staging: global -> regs -> ds_store, fragments via
//   ds_load in ISA WMMA layouts. One __syncthreads per k-chunk.
// PHASE 1: S = sum(exp(l)) over rows<nN, cols>=nN
// PHASE 2: sum(l - log(exp(l)+S+1e-9)) over rows<nN, cols<nN, diag l=0
// ---------------------------------------------------------------------------
template <int PHASE>
__global__ __launch_bounds__(256) void crc_gemm(const _Float16* __restrict__ fH,
                                                const int* __restrict__ nNp,
                                                double* __restrict__ Sacc,
                                                double* __restrict__ Lacc) {
    const int nN = *nNp;
    const int bM = blockIdx.x >> 5;        // 64 M-tiles of 128 rows
    const int bN = blockIdx.x & 31;        // 32 N-tiles of 256 cols
    const int rowBase = bM * 128;
    const int colBase = bN * 256;

    // Block-uniform tile culling.
    if (PHASE == 1) {
        if (rowBase >= nN || colBase + 256 <= nN) return;
    } else {
        if (rowBase >= nN || colBase >= nN) return;
    }

    __shared__ _Float16 As[2][128 * LDS_STRIDE];   // 20 KB
    __shared__ _Float16 Bs[2][256 * LDS_STRIDE];   // 40 KB

    const int t     = threadIdx.x;
    const int wave  = t >> 5;
    const int lane  = t & 31;
    const int waveM = wave >> 2;           // 0..1
    const int waveN = wave & 3;            // 0..3
    const int r     = lane & 15;
    const int hi    = lane >> 4;

    v8f acc[4][4] = {};
    v8h aReg[2], bReg[4];

    // ---- staging helpers (16B chunks; thread t owns chunks t + i*256) ----
    auto loadGlobal = [&](int kk) {
        #pragma unroll
        for (int i = 0; i < 2; ++i) {
            const int c = t + i * 256;             // 512 A-chunks
            const int row = c >> 2, seg = c & 3;
            aReg[i] = *(const v8h*)(fH + (size_t)(rowBase + row) * KDIM + kk + seg * 8);
        }
        #pragma unroll
        for (int i = 0; i < 4; ++i) {
            const int c = t + i * 256;             // 1024 B-chunks
            const int row = c >> 2, seg = c & 3;
            bReg[i] = *(const v8h*)(fH + (size_t)(colBase + row) * KDIM + kk + seg * 8);
        }
    };
    auto storeLDS = [&](int buf) {
        #pragma unroll
        for (int i = 0; i < 2; ++i) {
            const int c = t + i * 256;
            const int row = c >> 2, seg = c & 3;
            *(v8h*)(&As[buf][row * LDS_STRIDE + seg * 8]) = aReg[i];
        }
        #pragma unroll
        for (int i = 0; i < 4; ++i) {
            const int c = t + i * 256;
            const int row = c >> 2, seg = c & 3;
            *(v8h*)(&Bs[buf][row * LDS_STRIDE + seg * 8]) = bReg[i];
        }
    };

    loadGlobal(0);
    storeLDS(0);

    #pragma unroll 2
    for (int it = 0; it < KDIM / 32; ++it) {
        __syncthreads();
        if (it + 1 < KDIM / 32) loadGlobal((it + 1) * 32);

        const int buf = it & 1;
        v16h a[4], b[4];
        #pragma unroll
        for (int m = 0; m < 4; ++m) {
            // A 16x32 frag: K runs [hi*8, +8) and [hi*8+16, +8)
            const _Float16* p = &As[buf][(waveM * 64 + m * 16 + r) * LDS_STRIDE + hi * 8];
            v8h lo  = *(const v8h*)(p);
            v8h hi8 = *(const v8h*)(p + 16);
            a[m] = __builtin_shufflevector(lo, hi8, 0, 1, 2, 3, 4, 5, 6, 7,
                                                    8, 9, 10, 11, 12, 13, 14, 15);
        }
        #pragma unroll
        for (int n = 0; n < 4; ++n) {
            // B 32x16 frag: K runs [hi*8, +8) and [hi*8+16, +8)
            const _Float16* p = &Bs[buf][(waveN * 64 + n * 16 + r) * LDS_STRIDE + hi * 8];
            v8h lo  = *(const v8h*)(p);
            v8h hi8 = *(const v8h*)(p + 16);
            b[n] = __builtin_shufflevector(lo, hi8, 0, 1, 2, 3, 4, 5, 6, 7,
                                                    8, 9, 10, 11, 12, 13, 14, 15);
        }
        #pragma unroll
        for (int m = 0; m < 4; ++m)
            #pragma unroll
            for (int n = 0; n < 4; ++n)
                acc[m][n] = __builtin_amdgcn_wmma_f32_16x16x32_f16(
                    false, a[m], false, b[n], (short)0, acc[m][n], false, false);

        if (it + 1 < KDIM / 32) storeLDS((it + 1) & 1);
    }

    float Sv = 0.f;
    if (PHASE == 2) Sv = (float)(*Sacc) + 1e-9f;

    const int rowBaseW = rowBase + waveM * 64;
    const int colBaseW = colBase + waveN * 64;
    float local = 0.f;
    #pragma unroll
    for (int m = 0; m < 4; ++m) {
        const int ibase = rowBaseW + m * 16 + hi * 8;   // D layout: M = reg + hi*8
        #pragma unroll
        for (int n = 0; n < 4; ++n) {
            const int j = colBaseW + n * 16 + r;        // D layout: N = lane%16
            #pragma unroll
            for (int rr = 0; rr < 8; ++rr) {
                const int i = ibase + rr;
                float l = acc[m][n][rr] * TEMP_INV;
                if (PHASE == 1) {
                    if (i < nN && j >= nN) local += __expf(l);
                } else {
                    if (i < nN && j < nN) {
                        if (i == j) l = 0.f;            // zeroed diagonal
                        local += l - __logf(__expf(l) + Sv);
                    }
                }
            }
        }
    }
    #pragma unroll
    for (int off = 16; off >= 1; off >>= 1) local += __shfl_xor(local, off, 32);
    if (lane == 0) {
        if (PHASE == 1) atomicAdd(Sacc, (double)local);
        else            atomicAdd(Lacc, (double)local);
    }
}

// ---------------------------------------------------------------------------
// Kernel 5: finalize scalar loss.
// ---------------------------------------------------------------------------
__global__ void crc_final(const double* __restrict__ Lacc,
                          const int* __restrict__ nNp,
                          float* __restrict__ out) {
    const double nN    = (double)(*nNp);
    const double count = nN * nN;
    out[0] = (float)(-(*Lacc) / count * 0.1);   // SCALE_BY_TEMPERATURE
}

extern "C" void kernel_launch(void* const* d_in, const int* in_sizes, int n_in,
                              void* d_out, int out_size, void* d_ws, size_t ws_size,
                              hipStream_t stream) {
    const float* F      = (const float*)d_in[0];
    const int*   labels = (const int*)d_in[1];
    float*       out    = (float*)d_out;

    char* ws = (char*)d_ws;
    double*    Sacc = (double*)(ws + 0);
    double*    Lacc = (double*)(ws + 8);
    int*       nNp  = (int*)(ws + 16);
    int*       perm = (int*)(ws + 64);
    _Float16*  fH   = (_Float16*)(ws + 65536);   // 8192x1024 f16 = 16 MB

    crc_scan <<<1, 1024, 0, stream>>>(labels, perm, nNp, Sacc, Lacc);
    crc_norm <<<BDIM, 256, 0, stream>>>(F, perm, fH);
    crc_gemm<1><<<2048, 256, 0, stream>>>(fH, nNp, Sacc, Lacc);
    crc_gemm<2><<<2048, 256, 0, stream>>>(fH, nNp, Sacc, Lacc);
    crc_final<<<1, 1, 0, stream>>>(Lacc, nNp, out);
}
